// Loss_31877247271387
// MI455X (gfx1250) — compile-verified, matrix operand order checked by reading, not run on hardware
//
#include <hip/hip_runtime.h>
#include <hip/hip_bf16.h>
#include <math.h>

// ---------------------------------------------------------------------------
// MI455X / gfx1250: wave32, WMMA. Streaming-reduction problem (~151 MB read
// once, ~6.5 us at 23.3 TB/s). B128 loads -> VGPRs, hardware v_log_f32 for
// the BCE logs (precise ocml logf would make the kernel compute-bound:
// ~160 VALU cycles vs ~113 HBM lane-cycles per 32B), WMMA-with-ones as a
// wide exact f32 accumulator, ds_swizzle XOR butterflies for the wave32
// reduce, LDS block reduce, f32 global atomics.
// ---------------------------------------------------------------------------

typedef __attribute__((ext_vector_type(16))) _Float16 v16h;
typedef __attribute__((ext_vector_type(8)))  float    v8f;
typedef __attribute__((ext_vector_type(2)))  float    v2f;

#define BLOCK   256          // 8 waves of 32
#define A_ITERS 8            // BCE kernel: 8 x float4 per thread = 32 elems

// ws layout (floats): [0] = sum(l*logp + (1-l)*log1p(-p))  (negative)
//                     [1] = correct count
//                     [2] = sum(factor * w^2)  (both tensors)

__global__ void init_ws_kernel(float* __restrict__ ws) {
    if (threadIdx.x < 4) ws[threadIdx.x] = 0.0f;
}

// One-DS-op wave32 XOR butterfly step: ds_swizzle group-of-32,
// offset = {xor_mask[14:10], or_mask[9:5]=0, and_mask[4:0]=0x1F}.
template <int MASK>
__device__ __forceinline__ float xor_add(float x) {
    int i = __builtin_bit_cast(int, x);
    int j = __builtin_amdgcn_ds_swizzle(i, (MASK << 10) | 0x1F);
    return x + __builtin_bit_cast(float, j);
}
__device__ __forceinline__ float wave_reduce_add(float x) {
    x = xor_add<16>(x);
    x = xor_add<8>(x);
    x = xor_add<4>(x);
    x = xor_add<2>(x);
    x = xor_add<1>(x);
    return x;
}

// Fast log with torch-style clamp: v_log_f32 gives -inf at 0, so
// fmaxf(.,-100) reproduces torch's BCE log clamping exactly.
__device__ __forceinline__ float clamped_log(float x) {
    return fmaxf(__logf(x), -100.0f);
}

// ---------------------------------------------------------------------------
// BCE + count. B = 2^23 => 1024 blocks x 256 thr x 8 iters x float4, every
// thread runs the identical full trip count (EXEC all-1s for WMMA).
// Count trick: predicates accumulate 2-per-f16-slot (values 0/1/2, exact),
// one v_wmma_f32_16x16x32_f16 with B=ones reduces 1024 predicates/wave
// exactly into the f32 C accumulator (row sums <= 64, integers).
// ---------------------------------------------------------------------------
__global__ void __launch_bounds__(BLOCK)
bce_count_kernel(const float* __restrict__ p, const float* __restrict__ lab,
                 float* __restrict__ ws)
{
    const int lane = threadIdx.x & 31;
    const int wv   = threadIdx.x >> 5;

    float lsum = 0.0f;
    v16h preds;
    #pragma unroll
    for (int i = 0; i < 16; ++i) preds[i] = (_Float16)0.0f;

    const size_t base = (size_t)blockIdx.x * (BLOCK * 4 * A_ITERS)
                      + (size_t)threadIdx.x * 4;
    #pragma unroll
    for (int it = 0; it < A_ITERS; ++it) {
        const size_t idx = base + (size_t)it * (BLOCK * 4);
        const float4 pv = *reinterpret_cast<const float4*>(p   + idx);
        const float4 lv = *reinterpret_cast<const float4*>(lab + idx);
        const float px[4] = {pv.x, pv.y, pv.z, pv.w};
        const float lx[4] = {lv.x, lv.y, lv.z, lv.w};
        #pragma unroll
        for (int j = 0; j < 4; ++j) {
            const float x = px[j];
            const float l = lx[j];
            const float lp  = clamped_log(x);          // log(p)    clamped
            const float l1p = clamped_log(1.0f - x);   // log(1-p)  clamped
            // l*lp + (1-l)*l1p  ==  l1p + l*(lp - l1p)
            lsum += fmaf(l, lp - l1p, l1p);
            const int slot = (it & 3) * 4 + j;         // 2 preds per slot
            preds[slot] += (fabsf(x - l) < 0.5f) ? (_Float16)1.0f
                                                 : (_Float16)0.0f;
        }
    }

    // --- exact count via WMMA: D = preds(16x32) * ones(32x16) + 0 ---
    v16h onesh;
    #pragma unroll
    for (int i = 0; i < 16; ++i) onesh[i] = (_Float16)1.0f;
    v8f c = {};
    c = __builtin_amdgcn_wmma_f32_16x16x32_f16(
            /*neg_a=*/false, preds, /*neg_b=*/false, onesh,
            /*c_mod=*/(short)0, c, /*reuse_a=*/false, /*reuse_b=*/false);
    float csum = c[0] + c[1] + c[2] + c[3] + c[4] + c[5] + c[6] + c[7];
    // row sums replicate across 16 columns -> true count = sum(D)/16

    lsum = wave_reduce_add(lsum);
    csum = wave_reduce_add(csum);

    __shared__ float sl[BLOCK / 32];
    __shared__ float sc[BLOCK / 32];
    if (lane == 0) { sl[wv] = lsum; sc[wv] = csum; }
    __syncthreads();
    if (threadIdx.x == 0) {
        float L = 0.0f, C = 0.0f;
        #pragma unroll
        for (int i = 0; i < BLOCK / 32; ++i) { L += sl[i]; C += sc[i]; }
        unsafeAtomicAdd(&ws[0], L);
        unsafeAtomicAdd(&ws[1], C * (1.0f / 16.0f));  // exact integer
    }
}

// ---------------------------------------------------------------------------
// factor * sum(w^2) via V_WMMA_F32_16X16X4_F32 with all-ones B:
// D = A*1 + C accumulates 64 f32 squares per wave per instruction at full
// f32 precision while the VALU only squares + computes addresses. n must be
// a multiple of gridDim*1024 (both tensor sizes are powers of two) so every
// wave runs a uniform trip count (EXEC all-1s).
// ---------------------------------------------------------------------------
__global__ void __launch_bounds__(BLOCK)
wsumsq_kernel(const float* __restrict__ w, unsigned long long n, float factor,
              float* __restrict__ acc)
{
    const int lane = threadIdx.x & 31;
    const int wv   = threadIdx.x >> 5;

    v8f c = {};
    const v2f bones = {1.0f, 1.0f};

    const size_t stride = (size_t)gridDim.x * (BLOCK * 4);
    for (size_t idx = (size_t)blockIdx.x * (BLOCK * 4) + (size_t)threadIdx.x * 4;
         idx < (size_t)n; idx += stride) {
        const float4 v = *reinterpret_cast<const float4*>(w + idx);
        v2f a0 = {v.x * v.x, v.y * v.y};
        v2f a1 = {v.z * v.z, v.w * v.w};
        c = __builtin_amdgcn_wmma_f32_16x16x4_f32(
                false, a0, false, bones, (short)0, c, false, false);
        c = __builtin_amdgcn_wmma_f32_16x16x4_f32(
                false, a1, false, bones, (short)0, c, false, false);
    }

    float s = c[0] + c[1] + c[2] + c[3] + c[4] + c[5] + c[6] + c[7];
    s = wave_reduce_add(s);

    __shared__ float sb[BLOCK / 32];
    if (lane == 0) sb[wv] = s;
    __syncthreads();
    if (threadIdx.x == 0) {
        float S = 0.0f;
        #pragma unroll
        for (int i = 0; i < BLOCK / 32; ++i) S += sb[i];
        unsafeAtomicAdd(acc, S * (factor * (1.0f / 16.0f)));  // /16: col replication
    }
}

__global__ void finalize_kernel(const float* __restrict__ ws,
                                float* __restrict__ out, float invB)
{
    if (threadIdx.x == 0) {
        const float cross = -ws[0] * invB;            // -mean(...)
        const float reg   =  ws[2] * (0.5f * invB);   // sum(f*||w||^2) / (2B)
        out[0] = cross + reg;
        out[1] = ws[1];                               // correct count
    }
}

extern "C" void kernel_launch(void* const* d_in, const int* in_sizes, int n_in,
                              void* d_out, int out_size, void* d_ws, size_t ws_size,
                              hipStream_t stream)
{
    const float* p   = (const float*)d_in[0];   // model_outputs
    const float* lab = (const float*)d_in[1];   // labels
    const float* w1  = (const float*)d_in[2];   // 4096*4096
    const float* w2  = (const float*)d_in[3];   // 4096*1024
    float* out = (float*)d_out;
    float* ws  = (float*)d_ws;

    const long long B  = in_sizes[0];           // 8388608
    const unsigned long long n1 = (unsigned long long)in_sizes[2];
    const unsigned long long n2 = (unsigned long long)in_sizes[3];

    init_ws_kernel<<<1, 32, 0, stream>>>(ws);

    const int bceBlocks = (int)(B / (BLOCK * 4 * A_ITERS));   // 1024
    bce_count_kernel<<<bceBlocks, BLOCK, 0, stream>>>(p, lab, ws);

    // 1024 blocks * 1024 floats = 2^20 per sweep: w1 -> 16 uniform trips,
    // w2 -> 4 uniform trips (both divide exactly).
    wsumsq_kernel<<<1024, BLOCK, 0, stream>>>(w1, n1, 0.01f,  &ws[2]);
    wsumsq_kernel<<<1024, BLOCK, 0, stream>>>(w2, n2, 0.001f, &ws[2]);

    finalize_kernel<<<1, 32, 0, stream>>>(ws, out, 1.0f / (float)B);
}
